// FNO_72456098283960
// MI455X (gfx1250) — matrix operand order repaired and checked by reading, not compiled, not run on hardware
//
#include <hip/hip_runtime.h>
#include <math.h>

typedef __attribute__((ext_vector_type(16))) _Float16 v16h;
typedef __attribute__((ext_vector_type(8)))  float    v8f;

#define WCH 20
#define NMODE 16

union VU { v16h v; _Float16 h[16]; };
union CU { v8f v; float f[8]; };

__device__ __forceinline__ float gelu_exact(float x) {
    return 0.5f * x * (1.0f + erff(x * 0.7071067811865475f));
}

// 16-bit A-matrix (16x32) lane/half -> K mapping: contiguous, K = 16*hi + p (matches B fills)
__device__ __forceinline__ int kOfA(int p, int hi) {
    return 16 * hi + p;
}

// ---------------------------------------------------------------- max(point_data)
__global__ void fno_max(const float* __restrict__ pd, float* __restrict__ mv, int S) {
    __shared__ float red[1024];
    float m = -INFINITY;
    for (int i = threadIdx.x; i < S; i += 1024) m = fmaxf(m, pd[i]);
    red[threadIdx.x] = m;
    __syncthreads();
    for (int off = 512; off > 0; off >>= 1) {
        if ((int)threadIdx.x < off) red[threadIdx.x] = fmaxf(red[threadIdx.x], red[threadIdx.x + off]);
        __syncthreads();
    }
    if (threadIdx.x == 0) mv[0] = red[0];
}

// ---------------------------------------------------------------- zero scratch
__global__ void fno_zero(float* __restrict__ p, int n) {
    int id = blockIdx.x * blockDim.x + threadIdx.x;
    if (id < n) p[id] = 0.0f;
}

// ---------------------------------------------------------------- fc0 lift: (B,S,2)->(B,20,S)
__global__ void fno_fc0(const float* __restrict__ x, const float* __restrict__ pd,
                        const float* __restrict__ maxv, const float* __restrict__ w,
                        const float* __restrict__ bvec, float* __restrict__ h, int S) {
    int id = blockIdx.x * blockDim.x + threadIdx.x;
    int b = id / S;
    int s = id - b * S;
    float g = pd[s] / maxv[0];
    float xv = x[id];
#pragma unroll
    for (int c = 0; c < WCH; ++c)
        h[((size_t)(b * WCH + c)) * S + s] = xv * w[c] + g * w[WCH + c] + bvec[c];
}

// ---------------------------------------------------------------- forward truncated DFT
// F_re[b,c,k] = sum_s h*cos(k th), F_im = -sum_s h*sin(k th); k<16
__global__ void fno_fwd_dft(const float* __restrict__ h, float* __restrict__ fre,
                            float* __restrict__ fim, int S) {
    __shared__ float cosL[NMODE * 257];   // pad 257 to avoid bank conflicts
    __shared__ float sinL[NMODE * 257];
    __shared__ float hsh[256];
    __shared__ float part[32 * 8];
    const int b = blockIdx.y;
    const int tid = threadIdx.x;
    const int s0 = blockIdx.x * 256;
    {
        float th = 6.283185307179586f * ((float)(s0 + tid) / (float)S);
        float c1, s1;
        __sincosf(th, &s1, &c1);
        float cr = 1.0f, sr = 0.0f;
#pragma unroll
        for (int k = 0; k < NMODE; ++k) {
            cosL[k * 257 + tid] = cr;
            sinL[k * 257 + tid] = sr;
            float cn = cr * c1 - sr * s1;
            float sn = cr * s1 + sr * c1;
            cr = cn; sr = sn;
        }
    }
    __syncthreads();
    const int j = tid & 31;      // 0..15 cos comp, 16..31 sin comp
    const int g = tid >> 5;      // wave index = reduction group
    const float* tp = (j < NMODE) ? &cosL[j * 257] : &sinL[(j - NMODE) * 257];
    for (int c = 0; c < WCH; ++c) {
        hsh[tid] = h[((size_t)(b * WCH + c)) * S + s0 + tid];
        __syncthreads();
        float acc = 0.0f;
        for (int i = g; i < 256; i += 8) acc += hsh[i] * tp[i];
        part[j * 8 + g] = acc;
        __syncthreads();
        if (tid < 32) {
            float sum = 0.0f;
#pragma unroll
            for (int g2 = 0; g2 < 8; ++g2) sum += part[tid * 8 + g2];
            if (tid < NMODE) atomicAdd(&fre[(size_t)(b * WCH + c) * NMODE + tid], sum);
            else             atomicAdd(&fim[(size_t)(b * WCH + c) * NMODE + (tid - NMODE)], -sum);
        }
        __syncthreads();
    }
}

// ---------------------------------------------------------------- complex mode mixing -> Gcoef
// Gcoef[b,o,j]: j<16 cos coeff ((k==0?1:2)*gre/S), j>=16 sin coeff (-2*gim/S)
__global__ void fno_mode_mix(const float* __restrict__ fre, const float* __restrict__ fim,
                             const float* __restrict__ swre, const float* __restrict__ swim,
                             float* __restrict__ gcoef, int S, int total) {
    int id = blockIdx.x * blockDim.x + threadIdx.x;
    if (id >= total) return;
    int k = id % NMODE;
    int o = (id / NMODE) % WCH;
    int b = id / (NMODE * WCH);
    float gre = 0.0f, gim = 0.0f;
    for (int i = 0; i < WCH; ++i) {
        float fr = fre[(size_t)(b * WCH + i) * NMODE + k];
        float fi = fim[(size_t)(b * WCH + i) * NMODE + k];
        float wr = swre[(size_t)(i * WCH + o) * NMODE + k];
        float wi = swim[(size_t)(i * WCH + o) * NMODE + k];
        gre += fr * wr - fi * wi;
        gim += fr * wi + fi * wr;
    }
    float invS = 1.0f / (float)S;
    gcoef[(size_t)(b * WCH + o) * 32 + k]          = ((k == 0) ? gre : 2.0f * gre) * invS;
    gcoef[(size_t)(b * WCH + o) * 32 + NMODE + k]  = -2.0f * gim * invS;
}

// ---------------------------------------------------------------- fused inverse-DFT + 1x1 conv (+gelu), in place
// h[b,o,s] = act( sum_j Gcoef[b,o,j]*trig[j,s] + sum_c ww[o,c]*h[b,c,s] + wb[o] )
__global__ void fno_spectral_pointwise(float* __restrict__ h, const float* __restrict__ gcoef,
                                       const float* __restrict__ ww, const float* __restrict__ wb,
                                       int S, int doGelu) {
    const int b = blockIdx.y;
    const int wave = threadIdx.x >> 5;
    const int lane = threadIdx.x & 31;
    const int n = lane & 15;
    const int hi = lane >> 4;
    const size_t hb = (size_t)b * WCH * S;

    VU awLo, awHi, agLo, agHi;
#pragma unroll
    for (int p = 0; p < 16; ++p) {
        int k = kOfA(p, hi);
        int o2 = 16 + n;
        awLo.h[p] = (_Float16)((k < WCH) ? ww[n * WCH + k] : 0.0f);
        awHi.h[p] = (_Float16)((o2 < WCH && k < WCH) ? ww[o2 * WCH + k] : 0.0f);
        agLo.h[p] = (_Float16)gcoef[(size_t)(b * WCH + n) * 32 + k];
        agHi.h[p] = (_Float16)((o2 < WCH) ? gcoef[(size_t)(b * WCH + o2) * 32 + k] : 0.0f);
    }

    const int sBase = (blockIdx.x * 8 + wave) * 256;   // 16 tiles of 16 per wave
    for (int t = 0; t < 16; ++t) {
        const int s0 = sBase + t * 16;
        // B matrix from h: K=channel, N=s position
        VU bh;
#pragma unroll
        for (int p = 0; p < 16; ++p) {
            int c = hi * 16 + p;
            bh.h[p] = (_Float16)((c < WCH) ? h[hb + (size_t)c * S + s0 + n] : 0.0f);
        }
        // B matrix trig basis: K<16 -> cos(K th), K>=16 -> sin((K-16) th)
        VU bt;
        float th = 6.283185307179586f * ((float)(s0 + n) / (float)S);
        float c1, s1;
        __sincosf(th, &s1, &c1);
        float cr = 1.0f, sr = 0.0f;
#pragma unroll
        for (int p = 0; p < 16; ++p) {
            bt.h[p] = (_Float16)(hi ? sr : cr);
            float cn = cr * c1 - sr * s1;
            float sn = cr * s1 + sr * c1;
            cr = cn; sr = sn;
        }
        // output channels 0..15
        v8f acc = {};
        acc = __builtin_amdgcn_wmma_f32_16x16x32_f16(false, agLo.v, false, bt.v, (short)0, acc, false, false);
        acc = __builtin_amdgcn_wmma_f32_16x16x32_f16(false, awLo.v, false, bh.v, (short)0, acc, false, false);
        CU cu; cu.v = acc;
#pragma unroll
        for (int r = 0; r < 8; ++r) {
            int o = r + 8 * hi;
            float v = cu.f[r] + wb[o];
            if (doGelu) v = gelu_exact(v);
            h[hb + (size_t)o * S + s0 + n] = v;
        }
        // output channels 16..19
        v8f acc2 = {};
        acc2 = __builtin_amdgcn_wmma_f32_16x16x32_f16(false, agHi.v, false, bt.v, (short)0, acc2, false, false);
        acc2 = __builtin_amdgcn_wmma_f32_16x16x32_f16(false, awHi.v, false, bh.v, (short)0, acc2, false, false);
        CU cu2; cu2.v = acc2;
#pragma unroll
        for (int r = 0; r < 8; ++r) {
            int o = 16 + r + 8 * hi;
            if (o < WCH) {
                float v = cu2.f[r] + wb[o];
                if (doGelu) v = gelu_exact(v);
                h[hb + (size_t)o * S + s0 + n] = v;
            }
        }
    }
}

// ---------------------------------------------------------------- fc1(20->128)+gelu, fc2(128->1)
__global__ void fno_head(const float* __restrict__ h, const float* __restrict__ fc1w,
                         const float* __restrict__ fc1b, const float* __restrict__ fc2w,
                         const float* __restrict__ fc2b, float* __restrict__ out, int S) {
    const int b = blockIdx.y;
    const int wave = threadIdx.x >> 5;
    const int lane = threadIdx.x & 31;
    const int n = lane & 15;
    const int hi = lane >> 4;
    const size_t hb = (size_t)b * WCH * S;

    VU a[8];
#pragma unroll
    for (int t = 0; t < 8; ++t) {
#pragma unroll
        for (int p = 0; p < 16; ++p) {
            int k = kOfA(p, hi);
            a[t].h[p] = (_Float16)((k < WCH) ? fc1w[k * 128 + t * 16 + n] : 0.0f);
        }
    }
    const float fc2b0 = fc2b[0];
    const int sBase = (blockIdx.x * 8 + wave) * 256;
    for (int t = 0; t < 16; ++t) {
        const int s0 = sBase + t * 16;
        VU bh;
#pragma unroll
        for (int p = 0; p < 16; ++p) {
            int c = hi * 16 + p;
            bh.h[p] = (_Float16)((c < WCH) ? h[hb + (size_t)c * S + s0 + n] : 0.0f);
        }
        float partial = 0.0f;
#pragma unroll
        for (int tt = 0; tt < 8; ++tt) {
            v8f acc = {};
            acc = __builtin_amdgcn_wmma_f32_16x16x32_f16(false, a[tt].v, false, bh.v, (short)0, acc, false, false);
            CU cu; cu.v = acc;
#pragma unroll
            for (int r = 0; r < 8; ++r) {
                int o = tt * 16 + r + 8 * hi;
                float z = gelu_exact(cu.f[r] + fc1b[o]);
                partial += z * fc2w[o];
            }
        }
        float tot = partial + __shfl_xor(partial, 16, 32);
        if (lane < 16) out[(size_t)b * S + s0 + n] = tot + fc2b0;
    }
}

extern "C" void kernel_launch(void* const* d_in, const int* in_sizes, int n_in,
                              void* d_out, int out_size, void* d_ws, size_t ws_size,
                              hipStream_t stream) {
    (void)n_in; (void)out_size; (void)ws_size;
    const float* x    = (const float*)d_in[0];
    const float* pd   = (const float*)d_in[1];
    const float* fc0w = (const float*)d_in[2];
    const float* fc0b = (const float*)d_in[3];
    const float* fc1w = (const float*)d_in[4];
    const float* fc1b = (const float*)d_in[5];
    const float* fc2w = (const float*)d_in[6];
    const float* fc2b = (const float*)d_in[7];
    const int S = in_sizes[1];
    const int B = in_sizes[0] / S;
    float* out = (float*)d_out;

    char* ws = (char*)d_ws;
    size_t hBytes = (size_t)B * WCH * S * sizeof(float);
    float* hbuf = (float*)ws;
    size_t off = (hBytes + 255) & ~(size_t)255;
    float* fre   = (float*)(ws + off);
    float* fim   = fre + (size_t)B * WCH * NMODE;
    float* gcoef = fim + (size_t)B * WCH * NMODE;
    float* maxv  = gcoef + (size_t)B * WCH * 32;

    fno_max<<<1, 1024, 0, stream>>>(pd, maxv, S);
    fno_fc0<<<(B * S) / 256, 256, 0, stream>>>(x, pd, maxv, fc0w, fc0b, hbuf, S);
    for (int L = 0; L < 4; ++L) {
        const float* swre = (const float*)d_in[8 + 4 * L];
        const float* swim = (const float*)d_in[9 + 4 * L];
        const float* ww   = (const float*)d_in[10 + 4 * L];
        const float* wb   = (const float*)d_in[11 + 4 * L];
        int fcount = 2 * B * WCH * NMODE;
        fno_zero<<<(fcount + 255) / 256, 256, 0, stream>>>(fre, fcount);
        fno_fwd_dft<<<dim3(S / 256, B), 256, 0, stream>>>(hbuf, fre, fim, S);
        int total = B * WCH * NMODE;
        fno_mode_mix<<<(total + 255) / 256, 256, 0, stream>>>(fre, fim, swre, swim, gcoef, S, total);
        fno_spectral_pointwise<<<dim3(S / 2048, B), 256, 0, stream>>>(hbuf, gcoef, ww, wb, S, (L < 3) ? 1 : 0);
    }
    fno_head<<<dim3(S / 2048, B), 256, 0, stream>>>(hbuf, fc1w, fc1b, fc2w, fc2b, out, S);
}
